// CustomConv2D_18648747999847
// MI455X (gfx1250) — matrix operand order repaired and checked by reading, not compiled, not run on hardware
//
#include <hip/hip_runtime.h>

typedef __attribute__((ext_vector_type(16))) __bf16 v16bf;
typedef __attribute__((ext_vector_type(8)))  __bf16 v8bf;
typedef __attribute__((ext_vector_type(2)))  __bf16 v2bf;
typedef __attribute__((ext_vector_type(8)))  float  v8f;

#define N_IMG   32
#define C_IN    64
#define H_IN    112
#define W_IN    112
#define O_CH    128
#define K_TOT   576     // C_IN * 9
#define KC      64      // K chunk staged in LDS (2 WMMA k-steps of 32)
#define NCHUNK  9       // 576 / 64
#define TOH     8
#define TOW     16
#define PIX     128     // TOH * TOW pixels per block
#define THREADS 256     // 8 waves (wave32)
#define RS      72      // LDS row stride in bf16 elems: 64 + 8 pad (16B-aligned rows,
                        // 36-dword stride -> 16 distinct banks across 16 rows)

// fp32 = hi(bf16) + lo(bf16); a*b ~= ah*bh + ah*bl + al*bh  (error ~2^-17)
struct bf16pair { __bf16 h; __bf16 l; };

__device__ __forceinline__ bf16pair split_bf16(float v) {
    bf16pair r;
    r.h = (__bf16)v;
    r.l = (__bf16)(v - (float)r.h);
    return r;
}

__device__ __forceinline__ v16bf frag16(const __bf16* p0, const __bf16* p1) {
    v8bf a = *(const v8bf*)p0;
    v8bf b = *(const v8bf*)p1;
    return __builtin_shufflevector(a, b, 0,1,2,3,4,5,6,7,8,9,10,11,12,13,14,15);
}

__global__ __launch_bounds__(THREADS)
void conv3x3_wmma_bf16split(const float* __restrict__ x,
                            const float* __restrict__ w,
                            const float* __restrict__ bias,
                            float* __restrict__ out)
{
    // A = weights [o][k], B = im2col [pix][k]; hi/lo bf16 copies. 4 x 18 KB = 72 KB
    __shared__ __attribute__((aligned(16))) __bf16 wh [O_CH * RS];
    __shared__ __attribute__((aligned(16))) __bf16 wlo[O_CH * RS];
    __shared__ __attribute__((aligned(16))) __bf16 bh [PIX * RS];
    __shared__ __attribute__((aligned(16))) __bf16 blo[PIX * RS];

    const int tid  = threadIdx.x;
    const int wave = tid >> 5;
    const int lane = tid & 31;
    const int half = lane >> 4;
    const int l16  = lane & 15;

    // wave tile: 2 M-blocks x 4 N-blocks of the 8x8 subtile grid
    const int mg = wave >> 1;   // 0..3 -> M blocks {2mg, 2mg+1}
    const int ng = wave & 1;    // 0..1 -> N blocks {4ng .. 4ng+3}

    // block -> (image, spatial tile): 14 x 7 = 98 tiles per image
    const int bid = blockIdx.x;
    const int img = bid / 98;
    const int t   = bid - img * 98;
    const int ohT = t / 7;
    const int owT = t - ohT * 7;
    const int ohBase = ohT * TOH;
    const int owBase = owT * TOW;

    // bias fragments: lane L, vgpr v -> o = mg*32 + mi*16 + v + 8*half
    float biasv[2][8];
#pragma unroll
    for (int mi = 0; mi < 2; ++mi)
#pragma unroll
        for (int v = 0; v < 8; ++v)
            biasv[mi][v] = bias[mg * 32 + mi * 16 + v + 8 * half];

    v8f acc[2][4] = {};

    const int mRow0 = mg * 32 + l16;           // A rows for this lane
    const int mRow1 = mRow0 + 16;

    for (int kb = 0; kb < NCHUNK; ++kb) {
        const int kBase = kb * KC;

        // ---- stage weight chunk: hi/lo bf16 pairs, k-contiguous per o-row.
        // consecutive tid -> consecutive kp -> contiguous 8B global reads per o-row
#pragma unroll 4
        for (int it = 0; it < (O_CH * KC / 2) / THREADS; ++it) {
            int p  = it * THREADS + tid;        // 0..4095
            int o  = p >> 5;                    // / 32
            int kp = p & 31;
            const float* g = w + o * K_TOT + kBase + 2 * kp;
            bf16pair s0 = split_bf16(g[0]);
            bf16pair s1 = split_bf16(g[1]);
            v2bf hp, lp;
            hp.x = s0.h; hp.y = s1.h;
            lp.x = s0.l; lp.y = s1.l;
            *(v2bf*)&wh [o * RS + 2 * kp] = hp;
            *(v2bf*)&wlo[o * RS + 2 * kp] = lp;
        }

        // ---- stage im2col chunk: hi/lo bf16, [pix][k] rows.
        // consecutive tid -> consecutive pix -> consecutive iw (coalesced reads);
        // LDS stores stride 36 dwords across rows -> conflict-free
#pragma unroll 2
        for (int it = 0; it < (KC / 2 * PIX) / THREADS; ++it) {
            int p    = it * THREADS + tid;      // 0..4095
            int kp   = p >> 7;                  // / 128
            int pix  = p & 127;
            int oh_l = pix >> 4;
            int ow_l = pix & 15;
            int oh   = ohBase + oh_l;
            int ow   = owBase + ow_l;
            v2bf hp, lp;
#pragma unroll
            for (int e = 0; e < 2; ++e) {
                int k  = kBase + 2 * kp + e;
                int c  = k / 9;
                int r  = k - c * 9;
                int kh = r / 3;
                int kw = r - kh * 3;
                int ih = oh + kh - 1;           // PADDING = 1
                int iw = ow + kw - 1;
                float xv = 0.0f;
                if ((unsigned)ih < (unsigned)H_IN && (unsigned)iw < (unsigned)W_IN)
                    xv = x[((img * C_IN + c) * H_IN + ih) * W_IN + iw];
                bf16pair s = split_bf16(xv);
                if (e == 0) { hp.x = s.h; lp.x = s.l; }
                else        { hp.y = s.h; lp.y = s.l; }
            }
            *(v2bf*)&bh [pix * RS + 2 * kp] = hp;
            *(v2bf*)&blo[pix * RS + 2 * kp] = lp;
        }

        __syncthreads();

        // ---- compute: 2 k-steps of K=32; per step: 8 A b128 + 16 B b128 + 24 WMMA
#pragma unroll
        for (int ks = 0; ks < KC / 32; ++ks) {
            const int k0 = ks * 32;
            // A fragments: lanes 0-15 need K k0+[0..7],[16..23]; lanes 16-31 the +8 runs
            const __bf16* pah0 = &wh [mRow0 * RS + k0 + half * 8];
            const __bf16* pal0 = &wlo[mRow0 * RS + k0 + half * 8];
            const __bf16* pah1 = &wh [mRow1 * RS + k0 + half * 8];
            const __bf16* pal1 = &wlo[mRow1 * RS + k0 + half * 8];
            v16bf ah[2], al[2];
            ah[0] = frag16(pah0, pah0 + 16);
            al[0] = frag16(pal0, pal0 + 16);
            ah[1] = frag16(pah1, pah1 + 16);
            al[1] = frag16(pal1, pal1 + 16);
#pragma unroll
            for (int nb = 0; nb < 4; ++nb) {
                const int pix = ng * 64 + nb * 16 + l16;
                // B fragment: lanes 0-15 need K k0+[0..7],[16..23]; lanes 16-31 the +8 runs
                const __bf16* pbh = &bh [pix * RS + k0 + half * 8];
                const __bf16* pbl = &blo[pix * RS + k0 + half * 8];
                v16bf bhf = frag16(pbh, pbh + 16);
                v16bf blf = frag16(pbl, pbl + 16);
                // two independent accumulation chains (mi = 0,1) per N-block
#pragma unroll
                for (int mi = 0; mi < 2; ++mi) {
                    acc[mi][nb] = __builtin_amdgcn_wmma_f32_16x16x32_bf16(
                        false, ah[mi], false, blf, (short)0, acc[mi][nb], false, false);
                    acc[mi][nb] = __builtin_amdgcn_wmma_f32_16x16x32_bf16(
                        false, al[mi], false, bhf, (short)0, acc[mi][nb], false, false);
                    acc[mi][nb] = __builtin_amdgcn_wmma_f32_16x16x32_bf16(
                        false, ah[mi], false, bhf, (short)0, acc[mi][nb], false, false);
                }
            }
        }

        __syncthreads();
    }

    // ---- epilogue: lane L, vgpr v -> o = mg*32 + mi*16 + v + 8*half;
    // pixel = ng*64 + nb*16 + l16 -> oh = ohBase + ng*4 + nb, ow = owBase + l16
    // lanes 0..15 hit consecutive ow -> coalesced 64B segments
#pragma unroll
    for (int nb = 0; nb < 4; ++nb) {
        const int oh = ohBase + ng * 4 + nb;
        const int ow = owBase + l16;
#pragma unroll
        for (int mi = 0; mi < 2; ++mi) {
#pragma unroll
            for (int v = 0; v < 8; ++v) {
                const int o = mg * 32 + mi * 16 + v + 8 * half;
                out[((img * O_CH + o) * H_IN + oh) * W_IN + ow] =
                    acc[mi][nb][v] + biasv[mi][v];
            }
        }
    }
}

extern "C" void kernel_launch(void* const* d_in, const int* in_sizes, int n_in,
                              void* d_out, int out_size, void* d_ws, size_t ws_size,
                              hipStream_t stream) {
    (void)in_sizes; (void)n_in; (void)d_ws; (void)ws_size; (void)out_size;
    const float* x    = (const float*)d_in[0];
    const float* wgt  = (const float*)d_in[1];
    const float* bias = (const float*)d_in[2];
    float* out = (float*)d_out;

    const int tilesPerImg = (H_IN / TOH) * (W_IN / TOW);  // 14*7 = 98
    dim3 grid(N_IMG * tilesPerImg);                        // 3136 blocks
    dim3 block(THREADS);
    conv3x3_wmma_bf16split<<<grid, block, 0, stream>>>(x, wgt, bias, out);
}
